// LoopConnectivityDecoder_43396349559254
// MI455X (gfx1250) — matrix operand order repaired and checked by reading, not compiled, not run on hardware
//
#include <hip/hip_runtime.h>

typedef _Float16 f16;
typedef __attribute__((ext_vector_type(16))) _Float16 v16h;
typedef __attribute__((ext_vector_type(2)))  _Float16 v2h;
typedef __attribute__((ext_vector_type(8)))  float    v8f;

#define NROWS 1536
#define EMBD  32
#define HDIM  64

// ---------------------------------------------------------------------------
// Kernel 1: WMMA projection.  Computes per (row-tile, col-tile, which):
//   which==0:  ap = E @ W1[:, :32]^T + b1     (f16, row-major N x 64)
//   which==1:  bm = E @ W1[:, 32:]^T          (f16, row-major N x 64)
// One wave (32 lanes) per 16x16 tile, single v_wmma_f32_16x16x32_f16 (K=32=EMB).
// ---------------------------------------------------------------------------
__global__ __launch_bounds__(32)
void proj_wmma_kernel(const float* __restrict__ E,
                      const float* __restrict__ W1,
                      const float* __restrict__ b1,
                      f16* __restrict__ ap,
                      f16* __restrict__ bm)
{
    const int ti    = blockIdx.x;   // row tile   0..95
    const int hj    = blockIdx.y;   // col tile   0..3   (HDIM/16)
    const int which = blockIdx.z;   // 0 -> ap, 1 -> bm
    const int lane  = threadIdx.x;  // 0..31
    const int g     = lane >> 4;    // lane-group (K-half select)
    const int n     = lane & 15;    // A: row-in-tile ; B/D: col-in-tile

    const int h0   = hj * 16;
    const int koff = which * EMBD;  // column offset into W1 (2*EMB wide)

    // A fragment: 16x32 f16, row M = lane&15.
    //   element e -> K = 16*(e>>3) + 8*g + (e&7)
    // B fragment: 32x16 f16 (K x N), col N = lane&15.
    //   element e -> K = 16*(e>>3) + 8*g + (e&7)
    v16h A, B;
    const int arow = ti * 16 + n;
#pragma unroll
    for (int e = 0; e < 16; ++e) {
        const int ka = 16 * (e >> 3) + 8 * g + (e & 7);
        A[e] = (f16)E[arow * EMBD + ka];
        const int kb = 16 * (e >> 3) + 8 * g + (e & 7);
        B[e] = (f16)W1[(h0 + n) * (2 * EMBD) + koff + kb];
    }

    v8f c = {};
    c = __builtin_amdgcn_wmma_f32_16x16x32_f16(
            /*neg_a=*/false, A, /*neg_b=*/false, B,
            /*c_mod=*/(short)0, c, /*reuse_a=*/false, /*reuse_b=*/false);

    const float bias = (which == 0) ? b1[h0 + n] : 0.0f;
    f16* __restrict__ dst = (which == 0) ? ap : bm;
    // D layout: VGPR r holds row M = 8*g + r, col N = lane&15.
#pragma unroll
    for (int r = 0; r < 8; ++r) {
        const int m = ti * 16 + 8 * g + r;
        dst[m * HDIM + h0 + n] = (f16)(c[r] + bias);
    }
}

// ---------------------------------------------------------------------------
// Kernel 2: pairwise scores on the upper triangle, mirrored write.
//   s[i,j] = sigmoid( sum_k relu(ap[i,k] + bm[j,k]) * W2[k] + b2 ),  i<j
//   out[i][j] = out[j][i] = s ; out[i][i] = 0
// 16x16 tile per block; blocks with bi>bj exit. Packed f16 inner loop:
//   v_pk_add_f16 + v_pk_max + v_dot2_f32_f16 (f32 accumulation).
// ---------------------------------------------------------------------------
__global__ __launch_bounds__(256)
void pair_score_kernel(const f16* __restrict__ ap,
                       const f16* __restrict__ bm,
                       const float* __restrict__ W2,
                       const float* __restrict__ b2,
                       float* __restrict__ out)
{
    const int bj = blockIdx.x;
    const int bi = blockIdx.y;
    if (bi > bj) return;                  // triangular: skip lower tiles

    const int t  = threadIdx.x;
    const int tx = t & 15;                // column within tile (j)
    const int ty = t >> 4;                // row within tile (i)

    __shared__ f16 aps[16][HDIM + 8];     // padded stride: 72 halves
    __shared__ f16 bms[16][HDIM + 8];
    __shared__ v2h wsh[HDIM / 2];

    // Cooperative tile load: 16*64 = 1024 halves each, 4 per thread.
    {
        const int base = t * 4;           // 0..1020
        const int r = base >> 6;          // /HDIM
        const int c = base & (HDIM - 1);
        const f16* ga = ap + (bi * 16 + r) * HDIM + c;
        const f16* gb = bm + (bj * 16 + r) * HDIM + c;
#pragma unroll
        for (int q = 0; q < 4; ++q) {
            aps[r][c + q] = ga[q];
            bms[r][c + q] = gb[q];
        }
        if (t < HDIM / 2) {
            v2h w; w[0] = (f16)W2[2 * t]; w[1] = (f16)W2[2 * t + 1];
            wsh[t] = w;
        }
    }
    __syncthreads();

    const int i = bi * 16 + ty;
    const int j = bj * 16 + tx;
    const bool diag_tile = (bi == bj);

    if (i == j) {
        out[i * NROWS + i] = 0.0f;
        return;
    }
    if (diag_tile && ty > tx) return;     // lower half covered by mirror

    float acc = b2[0];
    const v2h zero = {};
#pragma unroll
    for (int kk = 0; kk < HDIM / 2; ++kk) {
        v2h a2 = *(const v2h*)&aps[ty][2 * kk];
        v2h c2 = *(const v2h*)&bms[tx][2 * kk];
        v2h r  = a2 + c2;                            // v_pk_add_f16
        r = __builtin_elementwise_max(r, zero);      // v_pk_max (relu)
#if __has_builtin(__builtin_amdgcn_fdot2)
        acc = __builtin_amdgcn_fdot2(r, wsh[kk], acc, false);
#else
        acc += (float)r[0] * (float)wsh[kk][0] + (float)r[1] * (float)wsh[kk][1];
#endif
    }

    const float s = 1.0f / (1.0f + __expf(-acc));
    out[i * NROWS + j] = s;
    out[j * NROWS + i] = s;
}

// ---------------------------------------------------------------------------
extern "C" void kernel_launch(void* const* d_in, const int* in_sizes, int n_in,
                              void* d_out, int out_size, void* d_ws, size_t ws_size,
                              hipStream_t stream)
{
    const float* E  = (const float*)d_in[0];  // (1536, 32)
    const float* W1 = (const float*)d_in[1];  // (64, 64)
    const float* b1 = (const float*)d_in[2];  // (64,)
    const float* W2 = (const float*)d_in[3];  // (1, 64)
    const float* b2 = (const float*)d_in[4];  // (1,)
    float* out = (float*)d_out;               // (1536, 1536)

    f16* ap = (f16*)d_ws;                     // 1536*64 f16
    f16* bm = ap + NROWS * HDIM;              // 1536*64 f16  (total 384 KB)

    proj_wmma_kernel<<<dim3(NROWS / 16, HDIM / 16, 2), 32, 0, stream>>>(E, W1, b1, ap, bm);
    pair_score_kernel<<<dim3(NROWS / 16, NROWS / 16), 256, 0, stream>>>(ap, bm, W2, b2, out);
}
